// DeepVCP_31224412242851
// MI455X (gfx1250) — matrix operand order repaired and checked by reading, not compiled, not run on hardware
//
#include <hip/hip_runtime.h>
#include <hip/hip_bf16.h>
#include <math.h>

// ---------------------------------------------------------------------------
// DeepVCP pipeline for MI455X (gfx1250, wave32, WMMA).
//   B=4, C=6, N=16384, K=256, NPOINT=64, NSAMPLE=32, RADIUS=1, KNN_K=10
// Kernels:
//   1) mlp_score_kernel : WMMA f16->f32 MLP + score dot, per-point scores
//   2) topk_kernel      : iterative argmax top-256 (deterministic)
//   3) gather_kernel    : gather keypts -> ws + d_out
//   4) group_kernel     : top-32 neighbor grouping (register sort network)
//   5) knn_kernel       : fp32 WMMA 16x16x4 distance tiles fused with
//                         in-register top-10, sqrt + normalize
// ---------------------------------------------------------------------------

typedef _Float16 v16h __attribute__((ext_vector_type(16)));
typedef float    v8f  __attribute__((ext_vector_type(8)));
typedef float    v2f  __attribute__((ext_vector_type(2)));

#define NPTS   16384
#define CCH    6
#define KSEL   256
#define FLTBIG 3.4e38f

// A-fragment K index for 16-bit 16x32 A (ISA 7.12.2):
// lanes 0-15: v0..v3 -> K 0..7, v4..v7 -> K 16..23
// lanes16-31: v0..v3 -> K 8..15, v4..v7 -> K 24..31
__device__ __forceinline__ int kOfA16(int e, bool hi) {
  int k = (e < 8) ? e : e + 8;
  return k + (hi ? 8 : 0);
}

// ------------------------------- kernel 1 ----------------------------------
__global__ __launch_bounds__(256) void mlp_score_kernel(
    const float* __restrict__ src, const float* __restrict__ tgt,
    const float* __restrict__ W1, const float* __restrict__ b1,
    const float* __restrict__ W2, const float* __restrict__ b2,
    const float* __restrict__ wsc, float* __restrict__ scores) {
  __shared__ float hbuf[8][16 * 64];  // per-wave h tile (16 pts x 64)

  const int  wv   = threadIdx.x >> 5;
  const int  lane = threadIdx.x & 31;
  const bool hi   = lane >= 16;
  const int  col  = lane & 15;
  const int  arr  = blockIdx.x >> 7;                  // 0=src 1=tgt
  const int  tile = (blockIdx.x & 127) * 8 + wv;      // 0..1023
  const int  n0   = tile * 16;
  const float* pts = arr ? tgt : src;
  float* sc_out = scores + arr * NPTS;

  // W1 B-fragments: 32x16 f16 (K=6 padded to 32), 4 col tiles.
  // B layout: lanes 0-15 hold K 0..15 (elem e -> K=e), lanes 16-31 K 16..31.
  v16h w1f[4];
#pragma unroll
  for (int nt = 0; nt < 4; ++nt) {
#pragma unroll
    for (int e = 0; e < 16; ++e) {
      int k = (hi ? 16 : 0) + e;
      w1f[nt][e] = (k < CCH) ? (_Float16)W1[k * 64 + nt * 16 + col]
                             : (_Float16)0.0f;
    }
  }

  float rowacc[8];
#pragma unroll
  for (int r = 0; r < 8; ++r) rowacc[r] = 0.0f;

  for (int b = 0; b < 4; ++b) {
    // A fragment: X tile (16 points x 6 channels padded to K=32)
    v16h xa;
#pragma unroll
    for (int e = 0; e < 16; ++e) {
      int k = (hi ? 16 : 0) + e;
      xa[e] = (k < CCH) ? (_Float16)pts[(b * CCH + k) * NPTS + n0 + col]
                        : (_Float16)0.0f;
    }
    // GEMM1 + bias + relu -> hbuf (f32)
#pragma unroll
    for (int nt = 0; nt < 4; ++nt) {
      v8f c;
#pragma unroll
      for (int r = 0; r < 8; ++r) c[r] = 0.0f;
      c = __builtin_amdgcn_wmma_f32_16x16x32_f16(false, xa, false, w1f[nt],
                                                 (short)0, c, false, false);
#pragma unroll
      for (int r = 0; r < 8; ++r) {
        int m  = r + (hi ? 8 : 0);
        int cc = nt * 16 + col;
        hbuf[wv][m * 64 + cc] = fmaxf(c[r] + b1[cc], 0.0f);
      }
    }
    __builtin_amdgcn_wave_barrier();  // hbuf store->load within wave

    // GEMM2: (16x64) @ (64x128), 8 col tiles x 2 K-chunks
    v8f acc2[8];
#pragma unroll
    for (int nt = 0; nt < 8; ++nt)
#pragma unroll
      for (int r = 0; r < 8; ++r) acc2[nt][r] = 0.0f;

#pragma unroll
    for (int kc = 0; kc < 2; ++kc) {
      v16h ha;
#pragma unroll
      for (int e = 0; e < 16; ++e) {
        int k = kc * 32 + (hi ? 16 : 0) + e;
        ha[e] = (_Float16)hbuf[wv][col * 64 + k];
      }
#pragma unroll
      for (int nt = 0; nt < 8; ++nt) {
        v16h wb;
#pragma unroll
        for (int e = 0; e < 16; ++e) {
          int k = kc * 32 + (hi ? 16 : 0) + e;
          wb[e] = (_Float16)W2[k * 128 + nt * 16 + col];
        }
        acc2[nt] = __builtin_amdgcn_wmma_f32_16x16x32_f16(
            false, ha, false, wb, (short)0, acc2[nt], false, false);
      }
    }

    // relu + dot with w_score: lane holds one column per tile
    float sc[8];
#pragma unroll
    for (int r = 0; r < 8; ++r) sc[r] = 0.0f;
#pragma unroll
    for (int nt = 0; nt < 8; ++nt) {
      int   cc = nt * 16 + col;
      float w  = wsc[cc];
      float bv = b2[cc];
#pragma unroll
      for (int r = 0; r < 8; ++r)
        sc[r] += fmaxf(acc2[nt][r] + bv, 0.0f) * w;
    }
    // reduce across the 16 lanes of each half (rows r / r+8)
#pragma unroll
    for (int r = 0; r < 8; ++r) {
      float v = sc[r];
      v += __shfl_xor(v, 8, 32);
      v += __shfl_xor(v, 4, 32);
      v += __shfl_xor(v, 2, 32);
      v += __shfl_xor(v, 1, 32);
      rowacc[r] += v;
    }
    __builtin_amdgcn_wave_barrier();  // before hbuf rewrite next batch
  }

  if (lane == 0) {
#pragma unroll
    for (int r = 0; r < 8; ++r) sc_out[n0 + r] = rowacc[r] * 0.25f;
  }
  if (lane == 16) {
#pragma unroll
    for (int r = 0; r < 8; ++r) sc_out[n0 + 8 + r] = rowacc[r] * 0.25f;
  }
}

// ------------------------------- kernel 2 ----------------------------------
__global__ __launch_bounds__(256) void topk_kernel(float* __restrict__ scores,
                                                   int* __restrict__ idx_out) {
  float* s   = scores + blockIdx.x * NPTS;
  int*   out = idx_out + blockIdx.x * KSEL;
  __shared__ float sv[256];
  __shared__ int   si[256];
  const int t = threadIdx.x;

  for (int k = 0; k < KSEL; ++k) {
    float bv = -FLTBIG;
    int   bi = 0;
    for (int i = t; i < NPTS; i += 256) {
      float v = s[i];
      if (v > bv || (v == bv && i < bi)) { bv = v; bi = i; }
    }
    sv[t] = bv;
    si[t] = bi;
    __syncthreads();
    for (int o = 128; o > 0; o >>= 1) {
      if (t < o) {
        float ov = sv[t + o];
        int   oi = si[t + o];
        if (ov > sv[t] || (ov == sv[t] && oi < si[t])) { sv[t] = ov; si[t] = oi; }
      }
      __syncthreads();
    }
    if (t == 0) {
      out[k]   = si[0];
      s[si[0]] = -FLTBIG;  // remove; re-written by kernel 1 on each call
    }
    __syncthreads();
  }
}

// ------------------------------- kernel 3 ----------------------------------
__global__ __launch_bounds__(256) void gather_kernel(
    const float* __restrict__ src, const float* __restrict__ tgt,
    const int* __restrict__ idx_src, const int* __restrict__ idx_tgt,
    float* __restrict__ keyp_src, float* __restrict__ keyp_tgt,
    float* __restrict__ out_keypts) {
  int t = blockIdx.x * 256 + threadIdx.x;  // 2*4*256*6 = 12288
  if (t >= 2 * 4 * KSEL * CCH) return;
  int arr = t / (4 * KSEL * CCH);
  int r   = t % (4 * KSEL * CCH);
  int b   = r / (KSEL * CCH);
  int k   = (r / CCH) % KSEL;
  int c   = r % CCH;
  const float* pts = arr ? tgt : src;
  const int*   idx = arr ? idx_tgt : idx_src;
  float v = pts[(b * CCH + c) * NPTS + idx[k]];
  (arr ? keyp_tgt : keyp_src)[r] = v;
  if (!arr) out_keypts[r] = v;  // src_keypts output
}

// ------------------------------- kernel 4 ----------------------------------
__global__ __launch_bounds__(256) void group_kernel(
    const float* __restrict__ keyp_src, const float* __restrict__ keyp_tgt,
    float* __restrict__ out) {
  const int arr = blockIdx.x;
  const float* kp = arr ? keyp_tgt : keyp_src;
  float* og = out + 88064 + arr * 49152;  // src_grouped / tgt_grouped
  int t = threadIdx.x;
  int b = t >> 6, j = t & 63;  // NPOINT=64 centers

  const float* ctr = &kp[(b * KSEL + j) * CCH];
  float cx = ctr[0], cy = ctr[1], cz = ctr[2];

  float bd[32];
  int   bi[32];
#pragma unroll
  for (int i = 0; i < 32; ++i) { bd[i] = FLTBIG; bi[i] = 0; }

  for (int k = 0; k < KSEL; ++k) {
    const float* p = &kp[(b * KSEL + k) * CCH];
    float dx = p[0] - cx, dy = p[1] - cy, dz = p[2] - cz;
    float v  = dx * dx + dy * dy + dz * dz;
    int   vi = k;
    if (v < bd[31]) {
#pragma unroll
      for (int i = 0; i < 32; ++i) {
        bool  sw = v < bd[i];
        float tv = bd[i];
        int   ti = bi[i];
        bd[i] = sw ? v : tv;  bi[i] = sw ? vi : ti;
        v     = sw ? tv : v;  vi    = sw ? ti : vi;
      }
    }
  }
  int i0 = bi[0];
#pragma unroll
  for (int s = 0; s < 32; ++s) {
    int gi = (bd[s] <= 1.0f) ? bi[s] : i0;  // RADIUS^2 = 1
    const float* p = &kp[(b * KSEL + gi) * CCH];
    float* o = &og[((b * 64 + j) * 32 + s) * CCH];
    o[0] = p[0] - cx; o[1] = p[1] - cy; o[2] = p[2] - cz;
    o[3] = p[3];      o[4] = p[4];      o[5] = p[5];
  }
}

// ------------------------------- kernel 5 ----------------------------------
// fp32 WMMA (16x16x4) distance tiles fused with per-row top-10.
__global__ __launch_bounds__(128) void knn_kernel(
    const float* __restrict__ keyp_src, const float* __restrict__ tgt_pts,
    float* __restrict__ out_dist) {
  __shared__ float cand[4][16][4];    // per wave: x,y,z,|c|^2
  __shared__ float tstage[16][4];     // shared target chunk: x,y,z
  __shared__ float dt[4][16][17];     // per wave distance tile (padded)

  const int  wv    = threadIdx.x >> 5;
  const int  lane  = threadIdx.x & 31;
  const bool hi    = lane >= 16;
  const int  col   = lane & 15;
  const int  b     = blockIdx.x >> 5;
  const int  cbase = (blockIdx.x & 31) * 64 + wv * 16;

  if (lane < 16) {
    int   kk = cbase + lane;   // candidate in [0,2048)
    int   k = kk >> 3, g = kk & 7;
    float x = keyp_src[(b * KSEL + k) * CCH + 0] + 0.1f * (float)((g >> 2) & 1);
    float y = keyp_src[(b * KSEL + k) * CCH + 1] + 0.1f * (float)((g >> 1) & 1);
    float z = keyp_src[(b * KSEL + k) * CCH + 2] + 0.1f * (float)(g & 1);
    cand[wv][lane][0] = x;
    cand[wv][lane][1] = y;
    cand[wv][lane][2] = z;
    cand[wv][lane][3] = x * x + y * y + z * z;
  }
  float best[10];
#pragma unroll
  for (int i = 0; i < 10; ++i) best[i] = FLTBIG;
  __syncthreads();

  for (int ch = 0; ch < NPTS / 16; ++ch) {
    if (threadIdx.x < 48) {
      int c = threadIdx.x >> 4;
      int n = ch * 16 + (threadIdx.x & 15);
      tstage[threadIdx.x & 15][c] = tgt_pts[(b * CCH + c) * NPTS + n];
    }
    __syncthreads();

    // A: 16x4 f32 (lanes 0-15: K0,K1; lanes 16-31: K2,K3); B: 4x16 analogous
    v2f a, bb;
    a[0]  = hi ? cand[wv][col][2] : cand[wv][col][0];
    a[1]  = hi ? 0.0f : cand[wv][col][1];
    bb[0] = hi ? tstage[col][2] : tstage[col][0];
    bb[1] = hi ? 0.0f : tstage[col][1];
    v8f acc;
#pragma unroll
    for (int r = 0; r < 8; ++r) acc[r] = 0.0f;
    acc = __builtin_amdgcn_wmma_f32_16x16x4_f32(false, a, false, bb, (short)0,
                                                acc, false, false);
    float tx = tstage[col][0], ty = tstage[col][1], tz = tstage[col][2];
    float tn = tx * tx + ty * ty + tz * tz;
#pragma unroll
    for (int r = 0; r < 8; ++r) {
      int m = r + (hi ? 8 : 0);
      dt[wv][m][col] = cand[wv][m][3] + tn - 2.0f * acc[r];
    }
    __builtin_amdgcn_wave_barrier();

    if (lane < 16) {  // lane owns candidate row `lane`
#pragma unroll
      for (int j = 0; j < 16; ++j) {
        float v = dt[wv][lane][j];
        if (v < best[9]) {
#pragma unroll
          for (int i = 0; i < 10; ++i) {
            float lo = fminf(best[i], v);
            v        = fmaxf(best[i], v);
            best[i]  = lo;
          }
        }
      }
    }
    __syncthreads();
  }

  if (lane < 16) {
    float d[10], s = 0.0f;
#pragma unroll
    for (int i = 0; i < 10; ++i) {
      d[i] = sqrtf(fmaxf(best[i], 0.0f));
      s += d[i];
    }
    float inv = 1.0f / s;
    int   kk  = cbase + lane;
#pragma unroll
    for (int i = 0; i < 10; ++i)
      out_dist[(b * 2048 + kk) * 10 + i] = d[i] * inv;
  }
}

// ------------------------------- launch ------------------------------------
extern "C" void kernel_launch(void* const* d_in, const int* in_sizes, int n_in,
                              void* d_out, int out_size, void* d_ws,
                              size_t ws_size, hipStream_t stream) {
  (void)in_sizes; (void)n_in; (void)out_size; (void)ws_size;
  const float* src = (const float*)d_in[0];
  const float* tgt = (const float*)d_in[1];
  const float* W1  = (const float*)d_in[2];
  const float* b1  = (const float*)d_in[3];
  const float* W2  = (const float*)d_in[4];
  const float* b2  = (const float*)d_in[5];
  const float* wsc = (const float*)d_in[6];
  float* out = (float*)d_out;

  char*  ws       = (char*)d_ws;
  float* scores   = (float*)ws;               // 2*16384 f32
  int*   idx      = (int*)(ws + 131072);      // 2*256 int
  float* keyp_src = (float*)(ws + 133120);    // 4*256*6 f32
  float* keyp_tgt = (float*)(ws + 157696);    // 4*256*6 f32

  mlp_score_kernel<<<256, 256, 0, stream>>>(src, tgt, W1, b1, W2, b2, wsc,
                                            scores);
  topk_kernel<<<2, 256, 0, stream>>>(scores, idx);
  gather_kernel<<<48, 256, 0, stream>>>(src, tgt, idx, idx + 256, keyp_src,
                                        keyp_tgt, out);
  group_kernel<<<2, 256, 0, stream>>>(keyp_src, keyp_tgt, out);
  knn_kernel<<<128, 128, 0, stream>>>(keyp_src, tgt, out + 6144);
}
